// EchoStateNetwork_8521215115557
// MI455X (gfx1250) — matrix operand order repaired and checked by reading, not compiled, not run on hardware
//
#include <hip/hip_runtime.h>
#include <math.h>

// Problem dims (fixed by the reference)
#define B_DIM 16
#define S_DIM 1024
#define D_DIM 512
#define N_DIM 2048

// Padded LDS row stride for the W_res slab: 2048 + 8 elements (=4112 bytes).
// Row n starts at bank (n*4112/4) % 64 = 4n % 64, so the 16 rows touched by a
// half-wave's 16-byte reads cover banks 4n..4n+3 -> conflict-free.
#define LDSW (N_DIM + 8)

typedef __attribute__((ext_vector_type(16))) __bf16 v16bf;
typedef __attribute__((ext_vector_type(8)))  __bf16 bf16x8;
typedef __attribute__((ext_vector_type(8)))  float  v8f;

// ---------------------------------------------------------------------------
// WMMA operand builders (CDNA5 wave32, V_WMMA_F32_16X16X32_BF16).
//
// A matrix 16x32 (MxK), row-major source with row stride `ldr` (elements):
//   lane 0-15 : M=lane,    holds K = k..k+15
//   lane16-31 : M=lane-16, holds K = k+16..k+31
// => per lane: two contiguous 16-byte chunks at k0 and k0+8, k0 = k + (hi?16:0)
// ---------------------------------------------------------------------------
static __device__ inline v16bf load_a_frag(const __bf16* __restrict__ base,
                                           int ldr, int k, int lane) {
  const int m  = lane & 15;
  const int k0 = k + ((lane & 16) ? 16 : 0);
  const __bf16* p = base + (size_t)m * ldr + k0;
  bf16x8 lo = *(const bf16x8*)(p);
  bf16x8 hi = *(const bf16x8*)(p + 8);
  return __builtin_shufflevector(lo, hi, 0, 1, 2, 3, 4, 5, 6, 7,
                                 8, 9, 10, 11, 12, 13, 14, 15);
}

// B matrix 32x16 (KxN) where B[k][n] = W[n][k] (W row-major, row stride ldw):
//   lane 0-15 : N=lane,    holds K = k..k+15   (16 consecutive elements of W row n)
//   lane16-31 : N=lane-16, holds K = k+16..k+31
// => per lane: one contiguous 32-byte run of W's row n.
static __device__ inline v16bf load_b_frag(const __bf16* __restrict__ wbase,
                                           int ldw, int k, int lane) {
  const int n  = lane & 15;
  const int k0 = k + ((lane & 16) ? 16 : 0);
  const __bf16* p = wbase + (size_t)n * ldw + k0;
  bf16x8 lo = *(const bf16x8*)(p);
  bf16x8 hi = *(const bf16x8*)(p + 8);
  return __builtin_shufflevector(lo, hi, 0, 1, 2, 3, 4, 5, 6, 7,
                                 8, 9, 10, 11, 12, 13, 14, 15);
}

// Same B-fragment gather but from the LDS-resident padded W slab.
static __device__ inline v16bf load_b_frag_lds(const __bf16* lds_w,
                                               int k, int lane) {
  const int n  = lane & 15;
  const int k0 = k + ((lane & 16) ? 16 : 0);
  const __bf16* p = lds_w + n * LDSW + k0;
  bf16x8 lo = *(const bf16x8*)(p);
  bf16x8 hi = *(const bf16x8*)(p + 8);
  return __builtin_shufflevector(lo, hi, 0, 1, 2, 3, 4, 5, 6, 7,
                                 8, 9, 10, 11, 12, 13, 14, 15);
}

// ---------------------------------------------------------------------------
// fp32 -> bf16 conversion (grid-stride)
// ---------------------------------------------------------------------------
__global__ void esn_convert_bf16(const float* __restrict__ in,
                                 __bf16* __restrict__ out, int n) {
  int i = blockIdx.x * blockDim.x + threadIdx.x;
  const int stride = gridDim.x * blockDim.x;
  for (; i < n; i += stride) out[i] = (__bf16)in[i];
}

__global__ void esn_init_counter(unsigned* cnt) {
  if (threadIdx.x == 0) *cnt = 0u;
}

// ---------------------------------------------------------------------------
// Phase 1: drive[s][b][n] = x[b,s,:]·W_in[n,:] + fb[b,s,:]·W_fb[n,:]
// One wave per 16x16 output tile (16 batches of one timestep x 16 neurons),
// K = D = 512 -> 16+16 bf16 WMMAs per tile.
// ---------------------------------------------------------------------------
__global__ __launch_bounds__(256)
void esn_drive_kernel(const __bf16* __restrict__ xb,
                      const __bf16* __restrict__ fbb,
                      const __bf16* __restrict__ win,
                      const __bf16* __restrict__ wfb,
                      float* __restrict__ drive) {
  const int lane = threadIdx.x & 31;
  const int wave = threadIdx.x >> 5;
  const int tile = blockIdx.x * 8 + wave;     // S * (N/16) = 131072 tiles
  const int s  = tile >> 7;                   // tile / 128
  const int n0 = (tile & 127) << 4;

  const __bf16* xs   = xb  + (size_t)s * D_DIM;   // row m=b at stride S*D
  const __bf16* fbs  = fbb + (size_t)s * D_DIM;
  const __bf16* winb = win + (size_t)n0 * D_DIM;
  const __bf16* wfbb = wfb + (size_t)n0 * D_DIM;

  v8f c = {0.f, 0.f, 0.f, 0.f, 0.f, 0.f, 0.f, 0.f};
#pragma unroll 4
  for (int k = 0; k < D_DIM; k += 32) {
    v16bf a1 = load_a_frag(xs, S_DIM * D_DIM, k, lane);
    v16bf b1 = load_b_frag(winb, D_DIM, k, lane);
    c = __builtin_amdgcn_wmma_f32_16x16x32_bf16(false, a1, false, b1,
                                                (short)0, c, false, false);
    v16bf a2 = load_a_frag(fbs, S_DIM * D_DIM, k, lane);
    v16bf b2 = load_b_frag(wfbb, D_DIM, k, lane);
    c = __builtin_amdgcn_wmma_f32_16x16x32_bf16(false, a2, false, b2,
                                                (short)0, c, false, false);
  }

  // C/D layout: lane 0-15 -> N=lane, rows M=0..7 in v0..7; lane 16-31 -> M=8..15
  const int n     = n0 + (lane & 15);
  const int mbase = (lane & 16) ? 8 : 0;
#pragma unroll
  for (int j = 0; j < 8; ++j) {
    drive[((size_t)s * B_DIM + (mbase + j)) * N_DIM + n] = c[j];
  }
}

// ---------------------------------------------------------------------------
// Phase 2: persistent recurrent scan.
// 128 blocks x 128 threads. Block g owns neurons [16g, 16g+16).
// The block's 16-row W_res slab (64 KB) is preloaded into LDS once and served
// from there for all 1024 steps (saves ~8 GB of L2 reads and removes L2
// latency from the serial critical path; LDS/WGP = 320 KB so it fits easily).
// 4 waves split K=2048 (512 each, 16 WMMAs), partials reduced through LDS.
// Double-buffered bf16 state + global atomic grid barrier per step.
// ---------------------------------------------------------------------------
__global__ __launch_bounds__(128)
void esn_scan_kernel(const __bf16* __restrict__ wres,
                     const float* __restrict__ drive,
                     __bf16* __restrict__ st0,
                     __bf16* __restrict__ st1,
                     float* __restrict__ out,
                     unsigned* __restrict__ cnt) {
  const int lane  = threadIdx.x & 31;
  const int wave  = threadIdx.x >> 5;
  const int n0    = blockIdx.x << 4;
  const int kbase = wave * (N_DIM / 4);
  const unsigned nblk = gridDim.x;

  __shared__ __bf16 lds_w[16 * LDSW];   // padded W_res slab (65792 B)
  __shared__ float  red[4][256];        // [wave][m*16 + n_local]

  // Preload W_res rows n0..n0+15 into LDS (16-byte vector copies).
  {
    const __bf16* wrow = wres + (size_t)n0 * N_DIM;
    for (int i = threadIdx.x; i < 16 * (N_DIM / 8); i += 128) {
      const int r  = i >> 8;        // / 256 chunks per row
      const int ch = i & 255;
      *(bf16x8*)&lds_w[r * LDSW + ch * 8] =
          *(const bf16x8*)&wrow[(size_t)r * N_DIM + ch * 8];
    }
  }
  __syncthreads();

  for (int s = 0; s < S_DIM; ++s) {
    const __bf16* st  = (s & 1) ? st1 : st0;
    __bf16*       stn = (s & 1) ? st0 : st1;

    v8f c = {0.f, 0.f, 0.f, 0.f, 0.f, 0.f, 0.f, 0.f};
#pragma unroll 4
    for (int kk = 0; kk < N_DIM / 4; kk += 32) {
      const int k = kbase + kk;
      v16bf a = load_a_frag(st, N_DIM, k, lane);        // state rows (batch)
      v16bf b = load_b_frag_lds(lds_w, k, lane);        // W_res rows from LDS
      c = __builtin_amdgcn_wmma_f32_16x16x32_bf16(false, a, false, b,
                                                  (short)0, c, false, false);
    }

    // Stash partial C tile into LDS in [m][n_local] order.
    {
      const int nl    = lane & 15;
      const int mbase = (lane & 16) ? 8 : 0;
#pragma unroll
      for (int j = 0; j < 8; ++j) red[wave][(mbase + j) * 16 + nl] = c[j];
    }
    __syncthreads();

    // Reduce 4 partials, add drive, tanh, emit fp32 output + bf16 next state.
    for (int i = threadIdx.x; i < 256; i += 128) {
      float v = red[0][i] + red[1][i] + red[2][i] + red[3][i];
      const int m = i >> 4;
      const int n = n0 + (i & 15);
      v += drive[((size_t)s * B_DIM + m) * N_DIM + n];
      const float t = tanhf(v);
      out[((size_t)m * S_DIM + s) * N_DIM + n] = t;
      stn[(size_t)m * N_DIM + n] = (__bf16)t;
      // Warm the next step's drive slice (global_prefetch_b8).
      if (s + 1 < S_DIM)
        __builtin_prefetch(&drive[((size_t)(s + 1) * B_DIM + m) * N_DIM + n],
                           0, 3);
    }

    __threadfence();        // make stn stores device-visible (release side)
    __syncthreads();        // all threads of block done before arrival

    if (threadIdx.x == 0) {
      __hip_atomic_fetch_add(cnt, 1u, __ATOMIC_RELEASE,
                             __HIP_MEMORY_SCOPE_AGENT);
      const unsigned target = nblk * (unsigned)(s + 1);
      while (__hip_atomic_load(cnt, __ATOMIC_ACQUIRE,
                               __HIP_MEMORY_SCOPE_AGENT) < target) {
        __builtin_amdgcn_s_sleep(8);
      }
    }
    __syncthreads();
    // Acquire for every thread: invalidate near caches before reading the
    // state tiles written by other WGPs.
    __builtin_amdgcn_fence(__ATOMIC_ACQUIRE, "agent");
  }
}

// ---------------------------------------------------------------------------
// Host-side launcher
// ---------------------------------------------------------------------------
extern "C" void kernel_launch(void* const* d_in, const int* in_sizes, int n_in,
                              void* d_out, int out_size, void* d_ws, size_t ws_size,
                              hipStream_t stream) {
  (void)in_sizes; (void)n_in; (void)out_size; (void)ws_size;

  const float* x_seq  = (const float*)d_in[0];   // (B,S,D)
  const float* fb_seq = (const float*)d_in[1];   // (B,S,D)
  const float* state0 = (const float*)d_in[2];   // (B,N) zeros
  const float* w_res  = (const float*)d_in[3];   // (N,N)
  const float* w_in   = (const float*)d_in[4];   // (N,D)
  const float* w_fb   = (const float*)d_in[5];   // (N,D)
  float* out = (float*)d_out;                    // (B,S,N)

  // Workspace layout (bytes): all chunks 16B-aligned.
  char* w = (char*)d_ws;
  unsigned* cnt = (unsigned*)w;                    w += 256;
  __bf16* wres_b = (__bf16*)w;                     w += (size_t)N_DIM * N_DIM * 2;
  __bf16* win_b  = (__bf16*)w;                     w += (size_t)N_DIM * D_DIM * 2;
  __bf16* wfb_b  = (__bf16*)w;                     w += (size_t)N_DIM * D_DIM * 2;
  __bf16* x_b    = (__bf16*)w;                     w += (size_t)B_DIM * S_DIM * D_DIM * 2;
  __bf16* fb_b   = (__bf16*)w;                     w += (size_t)B_DIM * S_DIM * D_DIM * 2;
  __bf16* st0    = (__bf16*)w;                     w += (size_t)B_DIM * N_DIM * 2;
  __bf16* st1    = (__bf16*)w;                     w += (size_t)B_DIM * N_DIM * 2;
  float*  drive  = (float*)w;                      // S*B*N fp32 = 134 MB

  // Phase 0: reset barrier counter; convert operands to bf16.
  esn_init_counter<<<1, 32, 0, stream>>>(cnt);
  esn_convert_bf16<<<2048, 256, 0, stream>>>(w_res, wres_b, N_DIM * N_DIM);
  esn_convert_bf16<<<1024, 256, 0, stream>>>(w_in,  win_b,  N_DIM * D_DIM);
  esn_convert_bf16<<<1024, 256, 0, stream>>>(w_fb,  wfb_b,  N_DIM * D_DIM);
  esn_convert_bf16<<<2048, 256, 0, stream>>>(x_seq,  x_b,  B_DIM * S_DIM * D_DIM);
  esn_convert_bf16<<<2048, 256, 0, stream>>>(fb_seq, fb_b, B_DIM * S_DIM * D_DIM);
  esn_convert_bf16<<<64,   256, 0, stream>>>(state0, st0,  B_DIM * N_DIM);

  // Phase 1: drive = x @ W_in^T + fb @ W_fb^T, laid out (S,B,N).
  // 131072 tiles / 8 waves per block.
  esn_drive_kernel<<<16384, 256, 0, stream>>>(x_b, fb_b, win_b, wfb_b, drive);

  // Phase 2: persistent serial scan over S=1024 steps.
  esn_scan_kernel<<<128, 128, 0, stream>>>(wres_b, drive, st0, st1, out, cnt);
}
